// BasicConvolutionBlock_32143535243816
// MI455X (gfx1250) — matrix operand order repaired and checked by reading, not compiled, not run on hardware
//
#include <hip/hip_runtime.h>
#include <hip/hip_bf16.h>

// ---------------------------------------------------------------------------
// Sparse 3x3x3 conv (CIN=32 -> COUT=64, 27 taps, gathered neighbors) + fused
// BatchNorm statistics + ReLU epilogue, built around V_WMMA_F32_16X16X32_BF16
// (K=32 == CIN, so one WMMA consumes a full input-channel dot product).
// ---------------------------------------------------------------------------

typedef __attribute__((ext_vector_type(16))) __bf16 v16bf;
typedef __attribute__((ext_vector_type(8)))  float  v8f;

#define CIN   32
#define COUT  64
#define KTAPS 27
#define EPS_BN 1e-5f

__device__ __forceinline__ unsigned int f2bf_rne(float f) {
    unsigned int u = __float_as_uint(f);
    unsigned int r = u + 0x7FFFu + ((u >> 16) & 1u);   // round-to-nearest-even
    return r >> 16;
}

// --- 1) x: fp32 [N,32] -> bf16 [N,32], 8 elements per thread ----------------
__global__ void cvt_x_kernel(const float* __restrict__ x,
                             unsigned short* __restrict__ xb, long n8) {
    long i = (long)blockIdx.x * blockDim.x + threadIdx.x;
    if (i >= n8) return;
    const float4* px = (const float4*)x;
    float4 a = px[2 * i];
    float4 b = px[2 * i + 1];
    uint4 o;
    o.x = f2bf_rne(a.x) | (f2bf_rne(a.y) << 16);
    o.y = f2bf_rne(a.z) | (f2bf_rne(a.w) << 16);
    o.z = f2bf_rne(b.x) | (f2bf_rne(b.y) << 16);
    o.w = f2bf_rne(b.z) | (f2bf_rne(b.w) << 16);
    ((uint4*)xb)[i] = o;
}

// --- 2) weights fp32 [27][32][64] -> bf16 in per-lane WMMA B layout ---------
// wb[((k*4 + ot)*32 + lane)*16 + j] = W[k][16*(lane>>4) + j][ot*16 + (lane&15)]
__global__ void cvt_w_kernel(const float* __restrict__ w,
                             unsigned short* __restrict__ wb) {
    int id = blockIdx.x * blockDim.x + threadIdx.x;
    if (id >= KTAPS * 4 * 32 * 16) return;
    int j    = id & 15;
    int lane = (id >> 4) & 31;
    int ot   = (id >> 9) & 3;
    int k    = id >> 11;
    int half = lane >> 4;
    int m    = lane & 15;
    int c    = 16 * half + j;       // input channel (K dim of WMMA)
    int o    = ot * 16 + m;         // output channel (N dim of WMMA)
    wb[id] = (unsigned short)f2bf_rne(w[(k * CIN + c) * COUT + o]);
}

// --- 3) zero the global sum / sumsq accumulators (128 floats) ---------------
__global__ void zero_stats_kernel(float* __restrict__ p) {
    int t = threadIdx.x;
    if (t < 2 * COUT) p[t] = 0.0f;
}

union ATile { v16bf v; uint4 u[2]; };

// --- 4) main conv: one wave per 16-voxel tile, 27 taps x 4 bf16 WMMAs -------
__global__ void __launch_bounds__(256)
conv_wmma_kernel(const unsigned short* __restrict__ xb,      // bf16 [N][32]
                 const unsigned short* __restrict__ wb,      // bf16 [27*4][32][16]
                 const int* __restrict__ nbr_idx,            // [27][N]
                 const unsigned char* __restrict__ nbr_mask, // [27][N] (bool)
                 float* __restrict__ out,                    // [N][64] pre-BN
                 float* __restrict__ gsum,                   // [64]
                 float* __restrict__ gssq,                   // [64]
                 int N, int ntiles) {
    __shared__ float lsum[COUT];
    __shared__ float lssq[COUT];
    int t = threadIdx.x;
    if (t < COUT) { lsum[t] = 0.0f; lssq[t] = 0.0f; }
    __syncthreads();

    const int lane   = t & 31;
    const int m      = lane & 15;     // row (A) / column (B,C) within tile
    const int half   = lane >> 4;     // which K half this lane owns
    const int wavesPerBlock = blockDim.x >> 5;
    const int gwave  = blockIdx.x * wavesPerBlock + (t >> 5);
    const int nwaves = gridDim.x * wavesPerBlock;

    float s0 = 0.f, s1 = 0.f, s2 = 0.f, s3 = 0.f;     // per-channel sums
    float q0 = 0.f, q1 = 0.f, q2 = 0.f, q3 = 0.f;     // per-channel sumsq

    for (int tile = gwave; tile < ntiles; tile += nwaves) {
        v8f acc0 = {}, acc1 = {}, acc2 = {}, acc3 = {};
        const int row_out = tile * 16 + m;
        const bool rvalid = row_out < N;

        for (int k = 0; k < KTAPS; ++k) {
            int src = 0, mk = 0;
            if (rvalid) {
                src = nbr_idx [(size_t)k * N + row_out];
                mk  = nbr_mask[(size_t)k * N + row_out];
            }
            // A: 16x32 bf16, contiguous layout -> lane holds K [16*half..+15]
            // of row m (two b128 gathers).
            const uint4* px = (const uint4*)(xb + (size_t)src * CIN);
            uint4 c0 = px[2 * half];
            uint4 c1 = px[2 * half + 1];
            if (!mk) { c0 = make_uint4(0, 0, 0, 0); c1 = c0; }
            ATile a; a.u[0] = c0; a.u[1] = c1;

            const unsigned short* wk = wb + ((size_t)k * 4 * 32) * 16;
            v16bf b0 = *(const v16bf*)(wk + (0 * 32 + lane) * 16);
            v16bf b1 = *(const v16bf*)(wk + (1 * 32 + lane) * 16);
            v16bf b2 = *(const v16bf*)(wk + (2 * 32 + lane) * 16);
            v16bf b3 = *(const v16bf*)(wk + (3 * 32 + lane) * 16);

            acc0 = __builtin_amdgcn_wmma_f32_16x16x32_bf16(false, a.v, false, b0,
                                                           (short)0, acc0, false, false);
            acc1 = __builtin_amdgcn_wmma_f32_16x16x32_bf16(false, a.v, false, b1,
                                                           (short)0, acc1, false, false);
            acc2 = __builtin_amdgcn_wmma_f32_16x16x32_bf16(false, a.v, false, b2,
                                                           (short)0, acc2, false, false);
            acc3 = __builtin_amdgcn_wmma_f32_16x16x32_bf16(false, a.v, false, b3,
                                                           (short)0, acc3, false, false);
        }

        // C layout: lane holds channel column (ot*16+m), rows M = 8*half + r.
        const int base_row = tile * 16 + 8 * half;
#pragma unroll
        for (int r = 0; r < 8; ++r) {
            float v0 = acc0[r], v1 = acc1[r], v2 = acc2[r], v3 = acc3[r];
            int rr = base_row + r;
            if (rr < N) {
                float* po = out + (size_t)rr * COUT + m;
                po[0]  = v0;
                po[16] = v1;
                po[32] = v2;
                po[48] = v3;
            }
            s0 += v0; q0 += v0 * v0;
            s1 += v1; q1 += v1 * v1;
            s2 += v2; q2 += v2 * v2;
            s3 += v3; q3 += v3 * v3;
        }
    }

    // fold the two wave halves (same channel, different M rows), then LDS.
    s0 += __shfl_xor(s0, 16, 32); q0 += __shfl_xor(q0, 16, 32);
    s1 += __shfl_xor(s1, 16, 32); q1 += __shfl_xor(q1, 16, 32);
    s2 += __shfl_xor(s2, 16, 32); q2 += __shfl_xor(q2, 16, 32);
    s3 += __shfl_xor(s3, 16, 32); q3 += __shfl_xor(q3, 16, 32);
    if (half == 0) {
        atomicAdd(&lsum[ 0 + m], s0); atomicAdd(&lssq[ 0 + m], q0);
        atomicAdd(&lsum[16 + m], s1); atomicAdd(&lssq[16 + m], q1);
        atomicAdd(&lsum[32 + m], s2); atomicAdd(&lssq[32 + m], q2);
        atomicAdd(&lsum[48 + m], s3); atomicAdd(&lssq[48 + m], q3);
    }
    __syncthreads();
    if (t < COUT) {
        atomicAdd(&gsum[t], lsum[t]);
        atomicAdd(&gssq[t], lssq[t]);
    }
}

// --- 5) finalize BN stats -> per-channel scale/shift ------------------------
__global__ void bn_stats_kernel(const float* __restrict__ gsum,
                                const float* __restrict__ gssq,
                                const float* __restrict__ gamma,
                                const float* __restrict__ beta,
                                float* __restrict__ scale,
                                float* __restrict__ shift, int N) {
    int c = threadIdx.x;
    if (c >= COUT) return;
    float inv  = 1.0f / (float)N;
    float mean = gsum[c] * inv;
    float var  = gssq[c] * inv - mean * mean;
    float sc   = gamma[c] * rsqrtf(var + EPS_BN);
    scale[c] = sc;
    shift[c] = beta[c] - mean * sc;
}

// --- 6) apply BN + ReLU in place on d_out, float4 vectorized ----------------
__global__ void bn_apply_kernel(float* __restrict__ out,
                                const float* __restrict__ scale,
                                const float* __restrict__ shift, long total4) {
    long i = (long)blockIdx.x * blockDim.x + threadIdx.x;
    if (i >= total4) return;
    float4 v = ((float4*)out)[i];
    int c = (int)((i * 4) & 63);
    v.x = fmaxf(fmaf(v.x, scale[c + 0], shift[c + 0]), 0.0f);
    v.y = fmaxf(fmaf(v.y, scale[c + 1], shift[c + 1]), 0.0f);
    v.z = fmaxf(fmaf(v.z, scale[c + 2], shift[c + 2]), 0.0f);
    v.w = fmaxf(fmaf(v.w, scale[c + 3], shift[c + 3]), 0.0f);
    ((float4*)out)[i] = v;
}

extern "C" void kernel_launch(void* const* d_in, const int* in_sizes, int n_in,
                              void* d_out, int out_size, void* d_ws, size_t ws_size,
                              hipStream_t stream) {
    const float*         x        = (const float*)d_in[0];
    const float*         w        = (const float*)d_in[1];
    const float*         gamma    = (const float*)d_in[2];
    const float*         beta     = (const float*)d_in[3];
    const int*           nbr_idx  = (const int*)d_in[4];
    const unsigned char* nbr_mask = (const unsigned char*)d_in[5]; // jnp.bool_ = 1B
    float*               out      = (float*)d_out;

    const int N = in_sizes[0] / CIN;

    // workspace layout
    char* ws = (char*)d_ws;
    size_t off = 0;
    unsigned short* xb = (unsigned short*)(ws + off);             // bf16 x
    off += (size_t)N * CIN * sizeof(unsigned short);
    off = (off + 255) & ~(size_t)255;
    unsigned short* wb = (unsigned short*)(ws + off);             // swizzled bf16 W
    off += (size_t)KTAPS * 4 * 32 * 16 * sizeof(unsigned short);
    off = (off + 255) & ~(size_t)255;
    float* stats = (float*)(ws + off);                            // sum[64] ++ ssq[64]
    off += 2 * COUT * sizeof(float);
    float* scale = (float*)(ws + off);
    off += COUT * sizeof(float);
    float* shift = (float*)(ws + off);
    float* gsum = stats;
    float* gssq = stats + COUT;

    // 1) convert x to bf16
    long n8 = (long)N * CIN / 8;
    cvt_x_kernel<<<(unsigned)((n8 + 255) / 256), 256, 0, stream>>>(x, xb, n8);
    // 2) swizzle weights into WMMA B layout
    cvt_w_kernel<<<(KTAPS * 4 * 32 * 16 + 255) / 256, 256, 0, stream>>>(w, wb);
    // 3) reset global stats (fresh every call -> graph-replay deterministic)
    zero_stats_kernel<<<1, 128, 0, stream>>>(stats);
    // 4) persistent WMMA conv with fused stats
    int ntiles = (N + 15) / 16;
    int blocks = 512;                       // 4096 waves, grid-stride over tiles
    conv_wmma_kernel<<<blocks, 256, 0, stream>>>(xb, wb, nbr_idx, nbr_mask,
                                                 out, gsum, gssq, N, ntiles);
    // 5) finalize stats
    bn_stats_kernel<<<1, 64, 0, stream>>>(gsum, gssq, gamma, beta, scale, shift, N);
    // 6) normalize + ReLU in place
    long total4 = (long)N * COUT / 4;
    bn_apply_kernel<<<(unsigned)((total4 + 255) / 256), 256, 0, stream>>>(out, scale,
                                                                          shift, total4);
}
